// HybridCNNLSTM_47459388621461
// MI455X (gfx1250) — compile-verified, matrix operand order checked by reading, not run on hardware
//
#include <hip/hip_runtime.h>
#include <hip/hip_bf16.h>

typedef __attribute__((ext_vector_type(16))) __bf16 v16bf;
typedef __attribute__((ext_vector_type(8)))  float  v8f;

__device__ __forceinline__ unsigned short f2bf(float x) {
  union { float f; unsigned u; } a; a.f = x;
  unsigned u = a.u;
  unsigned r = u + 0x7fffu + ((u >> 16) & 1u);   // round-to-nearest-even
  return (unsigned short)(r >> 16);
}

union HalfVec { v16bf v; unsigned short u[16]; };

__device__ __forceinline__ void cvt8(unsigned short* d, float4 a, float4 b) {
  d[0] = f2bf(a.x); d[1] = f2bf(a.y); d[2] = f2bf(a.z); d[3] = f2bf(a.w);
  d[4] = f2bf(b.x); d[5] = f2bf(b.y); d[6] = f2bf(b.z); d[7] = f2bf(b.w);
}

// gfx1250 async global->LDS copy (ASYNCcnt-tracked), 16 bytes per lane.
__device__ __forceinline__ void async_b128(unsigned lds_off, const void* g) {
  asm volatile("global_load_async_to_lds_b128 %0, %1, off"
               :: "v"(lds_off), "v"((unsigned long long)(size_t)g)
               : "memory");
}
__device__ __forceinline__ void wait_async0() {
  asm volatile("s_wait_asynccnt 0x0" ::: "memory");
}

// ---------------------------------------------------------------------------
// WMMA bf16 GEMM:  C[M,N] = op( A[M,K] @ W^T (+ bias) )
//   b_is_kn == 0: W stored [N,K] row-major (PyTorch linear weight, C = A@W.T)
//   b_is_kn == 1: W stored [K,N] row-major (C = A@W)
// Requires: K % 32 == 0, N % 64 == 0; M guarded at writeback (rows clamped on
// load -- an OOB A row only feeds an OOB C row, which is never stored).
// Block = 128 threads (4 waves). Block tile = 32(M) x 64(N); each wave owns a
// 16-col slice and two 16x16 accumulators -> 2 v_wmma per K-step.
// Tiles staged f32 into LDS via global_load_async_to_lds_b128.
// ---------------------------------------------------------------------------
__global__ void __launch_bounds__(128)
gemm_wmma_bf16(const float* __restrict__ A, const float* __restrict__ W,
               const float* __restrict__ bias, float* __restrict__ C,
               int M, int N, int K, int b_is_kn, int do_relu)
{
  __shared__ __align__(16) float As[32 * 32];   // [m][k]  4 KB
  __shared__ __align__(16) float Bs[64 * 32];   // mode0: [n][k]; mode1: [k][n]  8 KB

  const int tid  = threadIdx.x;
  const int lane = tid & 31;
  const int wv   = tid >> 5;
  const long m0  = (long)blockIdx.y * 32;
  const long n0  = (long)blockIdx.x * 64;
  const unsigned asbase = (unsigned)(size_t)(void*)As;
  const unsigned bsbase = (unsigned)(size_t)(void*)Bs;

  v8f acc0 = {0.f, 0.f, 0.f, 0.f, 0.f, 0.f, 0.f, 0.f};
  v8f acc1 = {0.f, 0.f, 0.f, 0.f, 0.f, 0.f, 0.f, 0.f};

  for (int kc = 0; kc < K; kc += 32) {
    // ---- async stage A tile (32x32 f32): 2 x 16B per thread ----
#pragma unroll
    for (int i = 0; i < 2; ++i) {
      int e4 = (tid + 128 * i) * 4;
      int m = e4 >> 5, kk = e4 & 31;
      long gm = m0 + m; if (gm > (long)M - 1) gm = (long)M - 1;  // clamp, no branchy load
      async_b128(asbase + e4 * 4, &A[gm * K + kc + kk]);
    }
    // ---- async stage B tile (64x32 f32): 4 x 16B per thread ----
#pragma unroll
    for (int i = 0; i < 4; ++i) {
      int e4 = (tid + 128 * i) * 4;
      const float* src;
      if (!b_is_kn) {
        int n = e4 >> 5, kk = e4 & 31;
        src = &W[(n0 + n) * K + kc + kk];          // contiguous along k
      } else {
        int n = e4 & 63, kk = e4 >> 6;
        src = &W[(long)(kc + kk) * N + n0 + n];    // contiguous along n
      }
      async_b128(bsbase + e4 * 4, src);
    }
    if (kc + 32 < K)  // global_prefetch for next A tile
      __builtin_prefetch(&A[(m0 + (tid & 31)) * K + kc + 32], 0, 1);

    wait_async0();
    __syncthreads();

    // ---- per-lane gather + f32->bf16 convert (CDNA5 16-bit WMMA layout) ----
    const int mr  = lane & 15;
    const int hi  = lane >> 4;
    const int ka4 = hi * 2;   // float4 index of K-block base within a 32-f32 row

    HalfVec a0, a1, bb;
    {
      const float4* r0 = (const float4*)&As[mr * 32];
      const float4* r1 = (const float4*)&As[(16 + mr) * 32];
      cvt8(a0.u + 0, r0[ka4],     r0[ka4 + 1]);
      cvt8(a0.u + 8, r0[4 + ka4], r0[5 + ka4]);
      cvt8(a1.u + 0, r1[ka4],     r1[ka4 + 1]);
      cvt8(a1.u + 8, r1[4 + ka4], r1[5 + ka4]);
    }
    const int nr = wv * 16 + (lane & 15);
    if (!b_is_kn) {
      const float4* rb = (const float4*)&Bs[nr * 32];
      cvt8(bb.u + 0, rb[ka4],     rb[ka4 + 1]);
      cvt8(bb.u + 8, rb[4 + ka4], rb[5 + ka4]);
    } else {
#pragma unroll
      for (int i2 = 0; i2 < 8; ++i2) {
        bb.u[i2]     = f2bf(Bs[(hi * 8 + i2) * 64 + nr]);
        bb.u[8 + i2] = f2bf(Bs[(16 + hi * 8 + i2) * 64 + nr]);
      }
    }
    acc0 = __builtin_amdgcn_wmma_f32_16x16x32_bf16(false, a0.v, false, bb.v,
                                                   (short)0, acc0, false, false);
    acc1 = __builtin_amdgcn_wmma_f32_16x16x32_bf16(false, a1.v, false, bb.v,
                                                   (short)0, acc1, false, false);
    __syncthreads();
  }

  // ---- writeback: VGPR r -> M = r (+8 upper lanes), N = lane&15 ----
  const long cn = n0 + wv * 16 + (lane & 15);
  const long mb = m0 + ((lane >> 4) ? 8 : 0);
  const float badd = bias ? bias[cn] : 0.f;
#pragma unroll
  for (int r = 0; r < 8; ++r) {
    long gm = mb + r;
    if (gm < M) {
      float v = acc0[r] + badd;
      if (do_relu) v = fmaxf(v, 0.f);
      C[gm * N + cn] = v;
    }
    long gm2 = mb + 16 + r;
    if (gm2 < M) {
      float v = acc1[r] + badd;
      if (do_relu) v = fmaxf(v, 0.f);
      C[gm2 * N + cn] = v;
    }
  }
}

// ---------------------------------------------------------------------------
// Conv1 (8->32, 3x3 SAME on 57x69) + ReLU + 2x2 VALID maxpool -> [192,32,28,34]
// ---------------------------------------------------------------------------
__global__ void conv1_relu_pool(const float* __restrict__ x, const float* __restrict__ w,
                                const float* __restrict__ b, float* __restrict__ out)
{
  int gid = blockIdx.x * blockDim.x + threadIdx.x;
  const int total = 192 * 32 * 28 * 34;
  if (gid >= total) return;
  int px = gid % 34;
  int py = (gid / 34) % 28;
  int oc = (gid / (34 * 28)) % 32;
  int f  = gid / (34 * 28 * 32);
  const float* xin = x + (long)f * 8 * 57 * 69;
  const float* wf  = w + oc * 8 * 9;
  float bb = b[oc];
  float mx = -1e30f;
#pragma unroll
  for (int dy = 0; dy < 2; ++dy)
#pragma unroll
    for (int dx = 0; dx < 2; ++dx) {
      int cy = py * 2 + dy, cx = px * 2 + dx;
      float s = bb;
      for (int ic = 0; ic < 8; ++ic) {
        const float* xc = xin + ic * 57 * 69;
        const float* wc = wf + ic * 9;
#pragma unroll
        for (int ky = 0; ky < 3; ++ky) {
          int iy = cy + ky - 1;
          if (iy < 0 || iy >= 57) continue;
#pragma unroll
          for (int kx = 0; kx < 3; ++kx) {
            int ix = cx + kx - 1;
            if (ix < 0 || ix >= 69) continue;
            s += xc[iy * 69 + ix] * wc[ky * 3 + kx];
          }
        }
      }
      mx = fmaxf(mx, s);
    }
  out[gid] = fmaxf(mx, 0.f);
}

// ---------------------------------------------------------------------------
// Conv2 (32->64, 3x3 SAME on 28x34) + ReLU + pool -> feats[192,15232]
// ---------------------------------------------------------------------------
__global__ void conv2_relu_pool(const float* __restrict__ x, const float* __restrict__ w,
                                const float* __restrict__ b, float* __restrict__ feats)
{
  int gid = blockIdx.x * blockDim.x + threadIdx.x;
  const int total = 192 * 64 * 14 * 17;
  if (gid >= total) return;
  int px = gid % 17;
  int py = (gid / 17) % 14;
  int oc = (gid / 238) % 64;
  int f  = gid / (238 * 64);
  const float* xin = x + (long)f * 32 * 28 * 34;
  const float* wf  = w + oc * 32 * 9;
  float bb = b[oc];
  float mx = -1e30f;
#pragma unroll
  for (int dy = 0; dy < 2; ++dy)
#pragma unroll
    for (int dx = 0; dx < 2; ++dx) {
      int cy = py * 2 + dy, cx = px * 2 + dx;
      float s = bb;
      for (int ic = 0; ic < 32; ++ic) {
        const float* xc = xin + ic * 28 * 34;
        const float* wc = wf + ic * 9;
#pragma unroll
        for (int ky = 0; ky < 3; ++ky) {
          int iy = cy + ky - 1;
          if (iy < 0 || iy >= 28) continue;
#pragma unroll
          for (int kx = 0; kx < 3; ++kx) {
            int ix = cx + kx - 1;
            if (ix < 0 || ix >= 34) continue;
            s += xc[iy * 34 + ix] * wc[ky * 3 + kx];
          }
        }
      }
      mx = fmaxf(mx, s);
    }
  feats[(long)f * 15232 + oc * 238 + py * 17 + px] = fmaxf(mx, 0.f);
}

// --------------------------- small utility kernels -------------------------
__global__ void zero_f(float* p, int n) {
  int i = blockIdx.x * blockDim.x + threadIdx.x;
  if (i < n) p[i] = 0.f;
}
__global__ void relu_f(float* p, int n) {
  int i = blockIdx.x * blockDim.x + threadIdx.x;
  if (i < n) p[i] = fmaxf(p[i], 0.f);
}

// degree over one batch graph (identical for all batches)
__global__ void degree_kernel(const int* __restrict__ edge, float* __restrict__ deg) {
  int e = blockIdx.x * blockDim.x + threadIdx.x;
  if (e >= 40000) return;
  int c = edge[40000 + e];
  c = min(max(c, 0), 4999);
  atomicAdd(&deg[c], 1.0f);
}
__global__ void dinv_kernel(const float* __restrict__ deg, float* __restrict__ dinv,
                            float* __restrict__ selfn) {
  int v = blockIdx.x * blockDim.x + threadIdx.x;
  if (v >= 5000) return;
  float di = rsqrtf(deg[v] + 1.0f);
  dinv[v] = di;
  selfn[v] = di * di;
}

// GCN layer 1: xw = z * w1 ; out = self_norm*xw + b1  (then edges scattered)
__global__ void gnn1_init(const float* __restrict__ zeta, const float* __restrict__ w1,
                          const float* __restrict__ b1, const float* __restrict__ selfn,
                          float* __restrict__ xw1, float* __restrict__ out1, int t)
{
  int gid = blockIdx.x * blockDim.x + threadIdx.x;
  if (gid >= 40000 * 32) return;
  int f  = gid & 31;
  int nd = gid >> 5;
  int v  = nd % 5000;
  int b  = nd / 5000;
  float z = zeta[(long)b * 24 * 5000 + (long)t * 5000 + v];
  float xw = z * w1[f];
  xw1[gid]  = xw;
  out1[gid] = selfn[v] * xw + b1[f];
}
__global__ void gnn1_scatter(const int* __restrict__ edge, const float* __restrict__ dinv,
                             const float* __restrict__ xw1, float* __restrict__ out1)
{
  int gid = blockIdx.x * blockDim.x + threadIdx.x;
  if (gid >= 8 * 40000 * 32) return;
  int f = gid & 31;
  int e = (gid >> 5) % 40000;
  int b = gid / (40000 * 32);
  int r = min(max(edge[e], 0), 4999);
  int c = min(max(edge[40000 + e], 0), 4999);
  float nm = dinv[r] * dinv[c];
  atomicAdd(&out1[((long)(b * 5000 + c)) * 32 + f], nm * xw1[((long)(b * 5000 + r)) * 32 + f]);
}

// GCN layer 2 propagate (transform done by WMMA GEMM)
__global__ void gnn2_init(const float* __restrict__ xw2, const float* __restrict__ b2,
                          const float* __restrict__ selfn, float* __restrict__ out2)
{
  int gid = blockIdx.x * blockDim.x + threadIdx.x;
  if (gid >= 40000 * 64) return;
  int f  = gid & 63;
  int nd = gid >> 6;
  int v  = nd % 5000;
  out2[gid] = selfn[v] * xw2[gid] + b2[f];
}
__global__ void gnn2_scatter(const int* __restrict__ edge, const float* __restrict__ dinv,
                             const float* __restrict__ xw2, float* __restrict__ out2)
{
  int gid = blockIdx.x * blockDim.x + threadIdx.x;
  if (gid >= 8 * 40000 * 64) return;
  int f = gid & 63;
  int e = (gid >> 6) % 40000;
  int b = gid / (40000 * 64);
  int r = min(max(edge[e], 0), 4999);
  int c = min(max(edge[40000 + e], 0), 4999);
  float nm = dinv[r] * dinv[c];
  atomicAdd(&out2[((long)(b * 5000 + c)) * 64 + f], nm * xw2[((long)(b * 5000 + r)) * 64 + f]);
}

// relu + mean over nodes -> node_summary[b][t][f]
__global__ void mean_relu_nodes(const float* __restrict__ out2, float* __restrict__ ns, int t)
{
  int bf = blockIdx.x;          // 0..511
  int b = bf >> 6, f = bf & 63;
  __shared__ float red[128];
  float s = 0.f;
  for (int v = threadIdx.x; v < 5000; v += 128)
    s += fmaxf(out2[((long)(b * 5000 + v)) * 64 + f], 0.f);
  red[threadIdx.x] = s;
  __syncthreads();
  for (int st = 64; st > 0; st >>= 1) {
    if (threadIdx.x < st) red[threadIdx.x] += red[threadIdx.x + st];
    __syncthreads();
  }
  if (threadIdx.x == 0) ns[(b * 24 + t) * 64 + f] = red[0] * (1.0f / 5000.0f);
}

// LSTM scan over T=24, B=8, HID=64. G = precomputed x@Wih.T, ordered [b*T+t][256]
__global__ void __launch_bounds__(256)
lstm_scan(const float* __restrict__ G, const float* __restrict__ whh,
          const float* __restrict__ bih, const float* __restrict__ bhh,
          float* __restrict__ hout)
{
  __shared__ float h[512], c[512], gates[256];
  int g = threadIdx.x;
  for (int i = g; i < 512; i += 256) { h[i] = 0.f; c[i] = 0.f; }
  __syncthreads();
  float bsum = bih[g] + bhh[g];
  for (int t = 0; t < 24; ++t) {
    for (int b = 0; b < 8; ++b) {
      float s = G[((long)(b * 24 + t)) * 256 + g] + bsum;
      const float* hb = &h[b * 64];
#pragma unroll 8
      for (int k = 0; k < 64; ++k) s += whh[g * 64 + k] * hb[k];
      gates[g] = s;
      __syncthreads();
      if (g < 64) {
        float ig = 1.f / (1.f + __expf(-gates[g]));
        float fg = 1.f / (1.f + __expf(-gates[64 + g]));
        float gg = tanhf(gates[128 + g]);
        float og = 1.f / (1.f + __expf(-gates[192 + g]));
        float cv = fg * c[b * 64 + g] + ig * gg;
        c[b * 64 + g] = cv;
        h[b * 64 + g] = og * tanhf(cv);
      }
      __syncthreads();
    }
  }
  for (int i = g; i < 512; i += 256) hout[i] = h[i];
}

__global__ void concat_k(const float* __restrict__ he, const float* __restrict__ hz,
                         float* __restrict__ comb)
{
  int i = blockIdx.x * blockDim.x + threadIdx.x;
  if (i >= 1024) return;
  int b = i >> 7, j = i & 127;
  comb[i] = (j < 64) ? he[b * 64 + j] : hz[b * 64 + (j - 64)];
}

// ---------------------------------------------------------------------------
extern "C" void kernel_launch(void* const* d_in, const int* in_sizes, int n_in,
                              void* d_out, int out_size, void* d_ws, size_t ws_size,
                              hipStream_t stream)
{
  (void)in_sizes; (void)n_in; (void)out_size; (void)ws_size;
  const float* era5 = (const float*)d_in[0];
  const float* zeta = (const float*)d_in[1];
  const int*   edge = (const int*)d_in[2];
  const float* c1w = (const float*)d_in[3];  const float* c1b = (const float*)d_in[4];
  const float* c2w = (const float*)d_in[5];  const float* c2b = (const float*)d_in[6];
  const float* ewih = (const float*)d_in[7]; const float* ewhh = (const float*)d_in[8];
  const float* ebih = (const float*)d_in[9]; const float* ebhh = (const float*)d_in[10];
  const float* g1w = (const float*)d_in[11]; const float* g1b = (const float*)d_in[12];
  const float* g2w = (const float*)d_in[13]; const float* g2b = (const float*)d_in[14];
  const float* zwih = (const float*)d_in[15]; const float* zwhh = (const float*)d_in[16];
  const float* zbih = (const float*)d_in[17]; const float* zbhh = (const float*)d_in[18];
  const float* f1w = (const float*)d_in[19]; const float* f1b = (const float*)d_in[20];
  const float* f2w = (const float*)d_in[21]; const float* f2b = (const float*)d_in[22];
  float* out = (float*)d_out;

  char* p = (char*)d_ws;
  auto alloc = [&](size_t nfl) -> float* {
    float* r = (float*)p;
    p += ((nfl * 4 + 255) / 256) * 256;
    return r;
  };
  float* pool1 = alloc(192UL * 32 * 28 * 34);   // 23.4 MB
  float* feats = alloc(192UL * 15232);          // 11.7 MB
  float* Ge    = alloc(192 * 256);
  float* Gz    = alloc(192 * 256);
  float* deg   = alloc(5000);
  float* dinv  = alloc(5000);
  float* selfn = alloc(5000);
  float* xw1   = alloc(40000UL * 32);           // 5.1 MB
  float* h1    = alloc(40000UL * 32);           // 5.1 MB
  float* xw2   = alloc(40000UL * 64);           // 10.2 MB
  float* h2b   = alloc(40000UL * 64);           // 10.2 MB
  float* ns    = alloc(8 * 24 * 64);
  float* he    = alloc(512);
  float* hz    = alloc(512);
  float* comb  = alloc(1024);
  float* fc1o  = alloc(8 * 512);

  const dim3 blk128(128);

  // --- ERA5 CNN branch ---
  conv1_relu_pool<<<(192 * 32 * 28 * 34 + 255) / 256, 256, 0, stream>>>(era5, c1w, c1b, pool1);
  conv2_relu_pool<<<(192 * 64 * 14 * 17 + 255) / 256, 256, 0, stream>>>(pool1, c2w, c2b, feats);
  // LSTM-e input projection: [192,15232] x [15232,256]^T -> Ge
  gemm_wmma_bf16<<<dim3(256 / 64, (192 + 31) / 32), blk128, 0, stream>>>(
      feats, ewih, nullptr, Ge, 192, 256, 15232, 0, 0);

  // --- graph normalization (identical per batch) ---
  zero_f<<<(5000 + 255) / 256, 256, 0, stream>>>(deg, 5000);
  degree_kernel<<<(40000 + 255) / 256, 256, 0, stream>>>(edge, deg);
  dinv_kernel<<<(5000 + 255) / 256, 256, 0, stream>>>(deg, dinv, selfn);

  // --- GCN per timestep ---
  for (int t = 0; t < 24; ++t) {
    gnn1_init<<<40000 * 32 / 256, 256, 0, stream>>>(zeta, g1w, g1b, selfn, xw1, h1, t);
    gnn1_scatter<<<8 * 40000 * 32 / 256, 256, 0, stream>>>(edge, dinv, xw1, h1);
    relu_f<<<40000 * 32 / 256, 256, 0, stream>>>(h1, 40000 * 32);
    // layer-2 transform: [40000,32] x [32,64] (K=32 == one WMMA step)
    gemm_wmma_bf16<<<dim3(1, 40000 / 32), blk128, 0, stream>>>(
        h1, g2w, nullptr, xw2, 40000, 64, 32, 1, 0);
    gnn2_init<<<40000 * 64 / 256, 256, 0, stream>>>(xw2, g2b, selfn, h2b);
    gnn2_scatter<<<8 * 40000 * 64 / 256, 256, 0, stream>>>(edge, dinv, xw2, h2b);
    mean_relu_nodes<<<512, 128, 0, stream>>>(h2b, ns, t);
  }

  // LSTM-z input projection: [192,64] x [64,256]^T -> Gz
  gemm_wmma_bf16<<<dim3(256 / 64, (192 + 31) / 32), blk128, 0, stream>>>(
      ns, zwih, nullptr, Gz, 192, 256, 64, 0, 0);

  // --- LSTM recurrences ---
  lstm_scan<<<1, 256, 0, stream>>>(Ge, ewhh, ebih, ebhh, he);
  lstm_scan<<<1, 256, 0, stream>>>(Gz, zwhh, zbih, zbhh, hz);

  // --- head ---
  concat_k<<<4, 256, 0, stream>>>(he, hz, comb);
  gemm_wmma_bf16<<<dim3(512 / 64, 1), blk128, 0, stream>>>(
      comb, f1w, f1b, fc1o, 8, 512, 128, 0, 1);
  gemm_wmma_bf16<<<dim3(120000 / 64, 1), blk128, 0, stream>>>(
      fc1o, f2w, f2b, out, 8, 120000, 512, 0, 0);
}
